// PoMMamba_85916525789301
// MI455X (gfx1250) — matrix operand order repaired and checked by reading, not compiled, not run on hardware
//
#include <hip/hip_runtime.h>
#include <hip/hip_bf16.h>
#include <math.h>

// ---- problem constants (from reference setup_inputs) ----
#define BB     2
#define NN     2048
#define DD     1024
#define DI     2048
#define DS     16
#define DTR    64
#define DCONV  4
#define KKEEP  1434           // ceil(2048 * 0.7)
#define EPSN   1e-5f

// register blocking for the WMMA GEMM: each wave computes (16*TM) x (16*TN)
#define TM 4
#define TN 2

typedef __attribute__((ext_vector_type(16))) _Float16 v16h;
typedef __attribute__((ext_vector_type(8)))  _Float16 v8h;
typedef __attribute__((ext_vector_type(8)))  float    v8f;

// ============================================================
// 1) residual add + RMSNorm -> res (f32), x (f16)
// ============================================================
__global__ void add_norm_kernel(const float* __restrict__ hs_in,
                                const float* __restrict__ res_in,
                                const float* __restrict__ norm_w,
                                float* __restrict__ res_out,
                                _Float16* __restrict__ x16) {
    const int tok = blockIdx.x;                 // 0 .. B*N-1
    const size_t base = (size_t)tok * DD;
    float r[4];
    float ss = 0.f;
    #pragma unroll
    for (int j = 0; j < 4; ++j) {
        const int c = threadIdx.x + j * 256;
        const float v = hs_in[base + c] + res_in[base + c];
        r[j] = v;
        res_out[base + c] = v;
        ss += v * v;
    }
    __shared__ float red[256];
    red[threadIdx.x] = ss;
    __syncthreads();
    for (int s = 128; s > 0; s >>= 1) {
        if (threadIdx.x < s) red[threadIdx.x] += red[threadIdx.x + s];
        __syncthreads();
    }
    const float scale = rsqrtf(red[0] * (1.0f / DD) + EPSN);
    #pragma unroll
    for (int j = 0; j < 4; ++j) {
        const int c = threadIdx.x + j * 256;
        x16[base + c] = (_Float16)(r[j] * scale * norm_w[c]);
    }
}

// ============================================================
// 2) generic f32 -> f16 convert
// ============================================================
__global__ void cvt_f16_kernel(const float* __restrict__ in,
                               _Float16* __restrict__ out, int n) {
    const int i = blockIdx.x * 256 + threadIdx.x;
    if (i < n) out[i] = (_Float16)in[i];
}

// strided: take first 64 of each 96-wide row (dt part of xdb)
__global__ void cvt_dtraw_kernel(const float* __restrict__ xdb,
                                 _Float16* __restrict__ dtraw) {
    const int i = blockIdx.x * 256 + threadIdx.x;   // B*N*64 threads
    const int row = i >> 6, c = i & 63;
    dtraw[i] = (_Float16)xdb[(size_t)row * (DTR + 2 * DS) + c];
}

// ============================================================
// 3) register-blocked WMMA f16 GEMM:
//    C[M,Nn] = A[M,K] * W[Nn,K]^T   (f32 accumulate)
//    each wave32 computes a 64x32 output block (TM x TN tiles of 16x16)
//    per K-step (32): 4 A-frags + 2 B-frags -> 8 v_wmma
//    mode 0: plain store.  mode 1: softplus(acc + bias[n]).
// ============================================================
__global__ void gemm_wmma_f16(const _Float16* __restrict__ A,
                              const _Float16* __restrict__ W,
                              float* __restrict__ C,
                              int M, int Nn, int K, int ldc,
                              const float* __restrict__ bias, int mode) {
    const int lane = threadIdx.x & 31;
    const int wave = threadIdx.x >> 5;
    const int wtile = blockIdx.x * (blockDim.x >> 5) + wave;
    const int tiles_n = Nn / (16 * TN);
    const int mt = wtile / tiles_n, nt = wtile % tiles_n;
    const int m0 = mt * (16 * TM), n0 = nt * (16 * TN);
    const int row = lane & 15;        // M row (A frags) / N col (B frags)
    const int h   = lane >> 4;        // lane half selects K sub-range

    // A fragment lane pointers: 16 contiguous halves at k = 16h of each row
    const _Float16* Ap[TM];
    #pragma unroll
    for (int i = 0; i < TM; ++i)
        Ap[i] = A + (size_t)(m0 + 16 * i + row) * K + 16 * h;
    // B fragment lane pointers: 16 contiguous halves at k = 16h of W[n,:]
    const _Float16* Bp[TN];
    #pragma unroll
    for (int j = 0; j < TN; ++j)
        Bp[j] = W + (size_t)(n0 + 16 * j + row) * K + 16 * h;

    v8f acc[TM][TN];
    #pragma unroll
    for (int i = 0; i < TM; ++i)
        #pragma unroll
        for (int j = 0; j < TN; ++j)
            acc[i][j] = (v8f){};

    for (int kb = 0; kb < K; kb += 32) {
        v16h a[TM], b[TN];
        #pragma unroll
        for (int i = 0; i < TM; ++i) {
            __builtin_prefetch(Ap[i] + kb + 128, 0, 0);   // global_prefetch_b8
            v8h lo = *(const v8h*)(Ap[i] + kb);
            v8h hi = *(const v8h*)(Ap[i] + kb + 8);
            a[i] = __builtin_shufflevector(lo, hi,
                                           0, 1, 2, 3, 4, 5, 6, 7,
                                           8, 9, 10, 11, 12, 13, 14, 15);
        }
        #pragma unroll
        for (int j = 0; j < TN; ++j) {
            __builtin_prefetch(Bp[j] + kb + 128, 0, 0);
            b[j] = *(const v16h*)(Bp[j] + kb);
        }
        #pragma unroll
        for (int i = 0; i < TM; ++i)
            #pragma unroll
            for (int j = 0; j < TN; ++j)
                acc[i][j] = __builtin_amdgcn_wmma_f32_16x16x32_f16(
                    /*neg_a=*/false, a[i], /*neg_b=*/false, b[j],
                    /*c_mod=*/(short)0, acc[i][j],
                    /*reuse_a=*/false, /*reuse_b=*/false);
    }

    #pragma unroll
    for (int j = 0; j < TN; ++j) {
        const int n = n0 + 16 * j + row;
        const float bv = (mode == 1) ? bias[n] : 0.f;
        #pragma unroll
        for (int i = 0; i < TM; ++i) {
            #pragma unroll
            for (int r = 0; r < 8; ++r) {
                const int m = m0 + 16 * i + r + 8 * h;
                float v = acc[i][j][r];
                if (mode == 1) {
                    v += bv;
                    v = (v > 20.f) ? v : log1pf(expf(v));   // softplus
                }
                C[(size_t)m * ldc + n] = v;
            }
        }
    }
}

// ============================================================
// 4) depthwise causal conv (width 4) + SiLU -> xc (f32 + f16)
// ============================================================
__global__ void conv_silu_kernel(const float* __restrict__ xz,   // (B,N,2*DI), xin = cols [0,DI)
                                 const float* __restrict__ conv_w,
                                 const float* __restrict__ conv_b,
                                 float* __restrict__ xc,
                                 _Float16* __restrict__ xc16) {
    const size_t idx = (size_t)blockIdx.x * 256 + threadIdx.x;   // B*N*DI
    const int d = (int)(idx % DI);
    const int t = (int)((idx / DI) % NN);
    const int b = (int)(idx / ((size_t)NN * DI));
    float acc = conv_b[d];
    #pragma unroll
    for (int j = 0; j < DCONV; ++j) {
        const int tt = t - (DCONV - 1) + j;
        if (tt >= 0)
            acc += xz[((size_t)(b * NN + tt)) * (2 * DI) + d] * conv_w[d * DCONV + j];
    }
    const float y = acc / (1.f + __expf(-acc));   // silu
    xc[idx] = y;
    xc16[idx] = (_Float16)y;
}

// ============================================================
// 5) selective scan over N, one lane per (b,d) channel.
//    B_t/C_t (32 floats) broadcast through LDS each step.
//    Fused epilogue: y = (y + D*xc) * silu(z), stored f16.
// ============================================================
__global__ void scan_kernel(const float* __restrict__ xdb,   // (B,N,96)
                            const float* __restrict__ dt,    // (B,N,DI)
                            const float* __restrict__ xc,    // (B,N,DI)
                            const float* __restrict__ xz,    // (B,N,2*DI): z = cols [DI,2*DI)
                            const float* __restrict__ A_log, // (DI,DS)
                            const float* __restrict__ D_param,
                            _Float16* __restrict__ y16) {
    const int b = blockIdx.y;
    const int d = blockIdx.x * 256 + threadIdx.x;

    float A[DS];
    #pragma unroll
    for (int s = 0; s < DS; ++s) A[s] = -__expf(A_log[d * DS + s]);
    const float Dp = D_param[d];

    float h[DS];
    #pragma unroll
    for (int s = 0; s < DS; ++s) h[s] = 0.f;

    __shared__ float sBC[2 * DS];   // [0,16) = B_t, [16,32) = C_t

    for (int t = 0; t < NN; ++t) {
        const size_t tok = (size_t)b * NN + t;
        if (threadIdx.x < 2 * DS)
            sBC[threadIdx.x] = xdb[tok * (DTR + 2 * DS) + DTR + threadIdx.x];
        __syncthreads();

        const float dtv = dt[tok * DI + d];
        const float xv  = xc[tok * DI + d];
        const float dx  = dtv * xv;
        float y = 0.f;
        #pragma unroll
        for (int s = 0; s < DS; ++s) {
            const float dA = __expf(dtv * A[s]);
            h[s] = dA * h[s] + dx * sBC[s];
            y += h[s] * sBC[DS + s];
        }
        __syncthreads();

        const float zv = xz[tok * (2 * DI) + DI + d];
        const float out = (y + Dp * xv) * (zv / (1.f + __expf(-zv)));
        y16[tok * DI + d] = (_Float16)out;
    }
}

// ============================================================
// 6) EViT: scores, stable ranks, rest softmax stats, gather+fuse
// ============================================================
__global__ void scores_kernel(const float* __restrict__ hs, float* __restrict__ scores) {
    const int tok = blockIdx.x;
    const size_t base = (size_t)tok * DD;
    float s = 0.f;
    #pragma unroll
    for (int j = 0; j < 4; ++j) {
        const float v = hs[base + threadIdx.x + j * 256];
        s += fmaxf(v, 0.f);
    }
    __shared__ float red[256];
    red[threadIdx.x] = s;
    __syncthreads();
    for (int w = 128; w > 0; w >>= 1) {
        if (threadIdx.x < w) red[threadIdx.x] += red[threadIdx.x + w];
        __syncthreads();
    }
    if (threadIdx.x == 0) {
        const float tot = red[0];
        scores[tok] = (tot == 0.f) ? -INFINITY : tot;
    }
}

// stable descending argsort via rank counting: order[rank] = t
__global__ void rank_kernel(const float* __restrict__ scores, int* __restrict__ order) {
    const int idx = blockIdx.x * 256 + threadIdx.x;   // B*N
    const int b = idx / NN, t = idx % NN;
    const float st = scores[idx];
    const float* sb = scores + (size_t)b * NN;
    int r = 0;
    for (int j = 0; j < NN; ++j) {
        const float sj = sb[j];
        r += (sj > st) || (sj == st && j < t);
    }
    order[(size_t)b * NN + r] = t;
}

// per batch: max & sum-exp over rest (indices [KKEEP,NN))
__global__ void rest_stats_kernel(const float* __restrict__ scores,
                                  const int* __restrict__ order,
                                  float* __restrict__ stats) {
    const int b = blockIdx.x;
    const float* sb = scores + (size_t)b * NN;
    const int* ob = order + (size_t)b * NN;
    __shared__ float red[256];
    float m = -INFINITY;
    for (int i = KKEEP + threadIdx.x; i < NN; i += 256) m = fmaxf(m, sb[ob[i]]);
    red[threadIdx.x] = m;
    __syncthreads();
    for (int w = 128; w > 0; w >>= 1) {
        if (threadIdx.x < w) red[threadIdx.x] = fmaxf(red[threadIdx.x], red[threadIdx.x + w]);
        __syncthreads();
    }
    const float mx = red[0];
    __syncthreads();
    float z = 0.f;
    for (int i = KKEEP + threadIdx.x; i < NN; i += 256) z += __expf(sb[ob[i]] - mx);
    red[threadIdx.x] = z;
    __syncthreads();
    for (int w = 128; w > 0; w >>= 1) {
        if (threadIdx.x < w) red[threadIdx.x] += red[threadIdx.x + w];
        __syncthreads();
    }
    if (threadIdx.x == 0) { stats[b * 2] = mx; stats[b * 2 + 1] = red[0]; }
}

__global__ void gather_kernel(const float* __restrict__ hs,
                              const float* __restrict__ res,
                              const int* __restrict__ order,
                              float* __restrict__ out_hs,
                              float* __restrict__ out_res) {
    const size_t idx = (size_t)blockIdx.x * 256 + threadIdx.x;   // B*KKEEP*DD
    const int d = (int)(idx % DD);
    const int i = (int)((idx / DD) % KKEEP);
    const int b = (int)(idx / ((size_t)DD * KKEEP));
    const int t = order[(size_t)b * NN + i];
    const size_t src = ((size_t)b * NN + t) * DD + d;
    const size_t dst = ((size_t)b * (KKEEP + 1) + i) * DD + d;
    out_hs[dst]  = hs[src];
    out_res[dst] = res[src];
}

__global__ void fuse_kernel(const float* __restrict__ hs,
                            const float* __restrict__ res,
                            const float* __restrict__ scores,
                            const int* __restrict__ order,
                            const float* __restrict__ stats,
                            float* __restrict__ out_hs,
                            float* __restrict__ out_res) {
    const int idx = blockIdx.x * 256 + threadIdx.x;   // B*DD
    const int b = idx / DD, d = idx % DD;
    const float m = stats[b * 2], invZ = 1.f / stats[b * 2 + 1];
    float ah = 0.f, ar = 0.f;
    for (int i = KKEEP; i < NN; ++i) {
        const int t = order[(size_t)b * NN + i];
        const float w = __expf(scores[(size_t)b * NN + t] - m) * invZ;
        const size_t src = ((size_t)b * NN + t) * DD + d;
        ah += w * hs[src];
        ar += w * res[src];
    }
    const size_t dst = ((size_t)b * (KKEEP + 1) + KKEEP) * DD + d;
    out_hs[dst]  = ah;
    out_res[dst] = ar;
}

// ============================================================
// host-side orchestration
// ============================================================
static inline size_t align256(size_t x) { return (x + 255) & ~(size_t)255; }

static inline void launch_gemm(const _Float16* A, const _Float16* W, float* C,
                               int M, int Nn, int K, int ldc,
                               const float* bias, int mode, hipStream_t stream) {
    const int waves = (M / (16 * TM)) * (Nn / (16 * TN));
    gemm_wmma_f16<<<waves / 8, 256, 0, stream>>>(A, W, C, M, Nn, K, ldc, bias, mode);
}

extern "C" void kernel_launch(void* const* d_in, const int* in_sizes, int n_in,
                              void* d_out, int out_size, void* d_ws, size_t ws_size,
                              hipStream_t stream) {
    const float* hs_in     = (const float*)d_in[0];
    const float* res_in    = (const float*)d_in[1];
    const float* norm_w    = (const float*)d_in[2];
    const float* in_proj_w = (const float*)d_in[3];
    const float* conv_w    = (const float*)d_in[4];
    const float* conv_b    = (const float*)d_in[5];
    const float* x_proj_w  = (const float*)d_in[6];
    const float* dt_proj_w = (const float*)d_in[7];
    const float* dt_proj_b = (const float*)d_in[8];
    const float* A_log     = (const float*)d_in[9];
    const float* D_param   = (const float*)d_in[10];
    const float* out_proj_w= (const float*)d_in[11];

    const size_t BND  = (size_t)BB * NN * DD;     // 4,194,304
    const size_t BNDI = (size_t)BB * NN * DI;     // 8,388,608

    // carve workspace
    char* p = (char*)d_ws;
    size_t off = 0;
    #define ALLOC(name, bytes) name = (void*)(p + off); off = align256(off + (bytes))
    void *w_res, *w_x16, *w_xz, *w_xc, *w_xc16, *w_xdb, *w_dtr16, *w_dt, *w_y16,
         *w_hs, *w_win16, *w_wx16, *w_wdt16, *w_wout16, *w_sc, *w_ord, *w_st;
    ALLOC(w_res,    BND * 4);
    ALLOC(w_x16,    BND * 2);
    ALLOC(w_xz,     (size_t)BB * NN * 2 * DI * 4);
    ALLOC(w_xc,     BNDI * 4);
    ALLOC(w_xc16,   BNDI * 2);
    ALLOC(w_xdb,    (size_t)BB * NN * (DTR + 2 * DS) * 4);
    ALLOC(w_dtr16,  (size_t)BB * NN * DTR * 2);
    ALLOC(w_dt,     BNDI * 4);
    ALLOC(w_y16,    BNDI * 2);
    ALLOC(w_hs,     BND * 4);
    ALLOC(w_win16,  (size_t)2 * DI * DD * 2);
    ALLOC(w_wx16,   (size_t)(DTR + 2 * DS) * DI * 2);
    ALLOC(w_wdt16,  (size_t)DI * DTR * 2);
    ALLOC(w_wout16, (size_t)DD * DI * 2);
    ALLOC(w_sc,     (size_t)BB * NN * 4);
    ALLOC(w_ord,    (size_t)BB * NN * 4);
    ALLOC(w_st,     (size_t)BB * 2 * 4);
    #undef ALLOC

    const int M = BB * NN;   // 4096 tokens

    // 1) residual add + rmsnorm
    add_norm_kernel<<<M, 256, 0, stream>>>(hs_in, res_in, norm_w,
                                           (float*)w_res, (_Float16*)w_x16);

    // 2) convert weights to f16
    {
        int n;
        n = 2 * DI * DD;
        cvt_f16_kernel<<<(n + 255) / 256, 256, 0, stream>>>(in_proj_w,  (_Float16*)w_win16, n);
        n = (DTR + 2 * DS) * DI;
        cvt_f16_kernel<<<(n + 255) / 256, 256, 0, stream>>>(x_proj_w,   (_Float16*)w_wx16, n);
        n = DI * DTR;
        cvt_f16_kernel<<<(n + 255) / 256, 256, 0, stream>>>(dt_proj_w,  (_Float16*)w_wdt16, n);
        n = DD * DI;
        cvt_f16_kernel<<<(n + 255) / 256, 256, 0, stream>>>(out_proj_w, (_Float16*)w_wout16, n);
    }

    // 3) GEMM1: xz = x @ in_proj_w.T  (4096 x 4096, K=1024)
    launch_gemm((const _Float16*)w_x16, (const _Float16*)w_win16, (float*)w_xz,
                M, 2 * DI, DD, 2 * DI, nullptr, 0, stream);

    // 4) depthwise conv + silu
    conv_silu_kernel<<<(unsigned)(BNDI / 256), 256, 0, stream>>>(
        (const float*)w_xz, conv_w, conv_b, (float*)w_xc, (_Float16*)w_xc16);

    // 5) GEMM2: xdb = xc @ x_proj_w.T  (4096 x 96, K=2048)
    launch_gemm((const _Float16*)w_xc16, (const _Float16*)w_wx16, (float*)w_xdb,
                M, DTR + 2 * DS, DI, DTR + 2 * DS, nullptr, 0, stream);

    // 6) extract dt columns as f16
    cvt_dtraw_kernel<<<(M * DTR) / 256, 256, 0, stream>>>((const float*)w_xdb,
                                                          (_Float16*)w_dtr16);

    // 7) GEMM3 + softplus: dt = softplus(dtraw @ dt_proj_w.T + b)  (4096 x 2048, K=64)
    launch_gemm((const _Float16*)w_dtr16, (const _Float16*)w_wdt16, (float*)w_dt,
                M, DI, DTR, DI, dt_proj_b, 1, stream);

    // 8) selective scan (fused gating epilogue)
    {
        dim3 g(DI / 256, BB);
        scan_kernel<<<g, 256, 0, stream>>>((const float*)w_xdb, (const float*)w_dt,
            (const float*)w_xc, (const float*)w_xz, A_log, D_param, (_Float16*)w_y16);
    }

    // 9) GEMM4: hs = y @ out_proj_w.T  (4096 x 1024, K=2048)
    launch_gemm((const _Float16*)w_y16, (const _Float16*)w_wout16, (float*)w_hs,
                M, DD, DI, DD, nullptr, 0, stream);

    // 10) EViT
    scores_kernel<<<M, 256, 0, stream>>>((const float*)w_hs, (float*)w_sc);
    rank_kernel<<<M / 256, 256, 0, stream>>>((const float*)w_sc, (int*)w_ord);
    rest_stats_kernel<<<BB, 256, 0, stream>>>((const float*)w_sc, (const int*)w_ord,
                                              (float*)w_st);

    float* out_hs  = (float*)d_out;
    float* out_res = out_hs + (size_t)BB * (KKEEP + 1) * DD;
    {
        const size_t n = (size_t)BB * KKEEP * DD;   // 2,936,832 (divisible by 256)
        gather_kernel<<<(unsigned)(n / 256), 256, 0, stream>>>((const float*)w_hs,
            (const float*)w_res, (const int*)w_ord, out_hs, out_res);
    }
    fuse_kernel<<<(BB * DD) / 256, 256, 0, stream>>>((const float*)w_hs,
        (const float*)w_res, (const float*)w_sc, (const int*)w_ord,
        (const float*)w_st, out_hs, out_res);
}
